// ConvLayer_23063974379610
// MI455X (gfx1250) — compile-verified, matrix operand order checked by reading, not run on hardware
//
#include <hip/hip_runtime.h>
#include <hip/hip_bf16.h>

#define KDIM   80
#define KP     96
#define N_FEAT 5
#define N_ROT  16
#define NS     512
#define NV     128
#define EPSF   1e-5f
#define TWO_PI_F 6.283185307179586f

typedef __attribute__((ext_vector_type(16))) _Float16 v16h;
typedef __attribute__((ext_vector_type(8)))  float    v8f;

__global__ __launch_bounds__(256) void conv_fused_kernel(
    const float* __restrict__ x,
    const float* __restrict__ mu_rho,
    const float* __restrict__ sigma_rho,
    const float* __restrict__ mu_theta,
    const float* __restrict__ sigma_theta,
    const float* __restrict__ W,
    const float* __restrict__ bconv,
    float* __restrict__ out)
{
    const int s    = blockIdx.x;
    const int tid  = threadIdx.x;
    const int wave = tid >> 5;   // 0..7
    const int lane = tid & 31;

    __shared__ float sh_mu_r[N_FEAT * KDIM];
    __shared__ float sh_ivr [N_FEAT * KDIM];
    __shared__ float sh_mu_t[N_FEAT * KDIM];
    __shared__ float sh_ivt [N_FEAT * KDIM];
    __shared__ float sh_desc[N_FEAT][N_ROT][KP];   // k padded 80->96 with zeros
    __shared__ float sh_x[NV * 8];

    // ---- load params (precompute 1/(sigma^2+eps)) and the sample's x slice ----
    for (int t = tid; t < N_FEAT * KDIM; t += 256) {
        sh_mu_r[t] = mu_rho[t];
        float sr = sigma_rho[t];
        sh_ivr[t] = 1.0f / (sr * sr + EPSF);
        sh_mu_t[t] = mu_theta[t];
        float st = sigma_theta[t];
        sh_ivt[t] = 1.0f / (st * st + EPSF);
    }
    for (int t = tid; t < N_FEAT * N_ROT * KP; t += 256)
        (&sh_desc[0][0][0])[t] = 0.0f;
    for (int t = tid; t < NV * 8; t += 256)
        sh_x[t] = x[(size_t)s * NV * 8 + t];
    __syncthreads();

    // ---- Phase 1: descriptor accumulation.  wave w owns rotations {w, w+8};
    //      lanes stripe k over 3x32 (k<80 valid). ----
    const float rot_step = TWO_PI_F / (float)N_ROT;

    for (int v = 0; v < NV; ++v) {
        const float* xv = &sh_x[v * 8];
        float f[N_FEAT];
#pragma unroll
        for (int i = 0; i < N_FEAT; ++i) f[i] = xv[i];
        const float rho = xv[5];
        const float th  = xv[6];
        const float mk  = xv[7];

        float grho[3][N_FEAT];
#pragma unroll
        for (int j = 0; j < 3; ++j) {
            int k = lane + 32 * j;
            bool valid = (k < KDIM);
#pragma unroll
            for (int i = 0; i < N_FEAT; ++i) {
                if (valid) {
                    int idx = i * KDIM + k;
                    float d = rho - sh_mu_r[idx];
                    grho[j][i] = __expf(-d * d * sh_ivr[idx]);
                } else {
                    grho[j][i] = 0.0f;
                }
            }
        }

#pragma unroll
        for (int rr = 0; rr < 2; ++rr) {
            int r = wave + rr * 8;
            float tr = th + (float)r * rot_step;      // < 4*pi
            if (tr >= TWO_PI_F) tr -= TWO_PI_F;       // mod 2*pi

            float g[3][N_FEAT];
            float sum[N_FEAT];
#pragma unroll
            for (int i = 0; i < N_FEAT; ++i) sum[i] = 0.0f;

#pragma unroll
            for (int j = 0; j < 3; ++j) {
                int k = lane + 32 * j;
                bool valid = (k < KDIM);
#pragma unroll
                for (int i = 0; i < N_FEAT; ++i) {
                    float gg = 0.0f;
                    if (valid) {
                        int idx = i * KDIM + k;
                        float d = tr - sh_mu_t[idx];
                        gg = grho[j][i] * __expf(-d * d * sh_ivt[idx]) * mk;
                    }
                    g[j][i] = gg;
                    sum[i] += gg;
                }
            }
            // wave32 reduction of the per-feature k-sums
#pragma unroll
            for (int off = 16; off > 0; off >>= 1) {
#pragma unroll
                for (int i = 0; i < N_FEAT; ++i)
                    sum[i] += __shfl_xor(sum[i], off, 32);
            }
            // normalized accumulate into desc (disjoint (i,r,k) cells per wave/lane)
#pragma unroll
            for (int j = 0; j < 3; ++j) {
                int k = lane + 32 * j;
                if (k < KDIM) {
#pragma unroll
                    for (int i = 0; i < N_FEAT; ++i)
                        sh_desc[i][r][k] += g[j][i] / (sum[i] + EPSF) * f[i];
                }
            }
        }
    }
    __syncthreads();

    // ---- Phase 2: per-feature WMMA GEMM conv[16xKDIM] = desc[16xKP] * W[KPxKDIM],
    //      then max over rotations (M dim), bias, ReLU, store. Waves 0..4 only
    //      (wave-uniform branch: EXEC stays all-1s inside). ----
    if (wave < N_FEAT) {
        const int i    = wave;
        const int M    = lane & 15;          // A: row M / B: column N / C: N
        const int koff = (lane >> 4) * 8;    // CDNA5 16-bit A/B half-lane K offset

        v8f acc[5];
#pragma unroll
        for (int nt = 0; nt < 5; ++nt)
            acc[nt] = (v8f){0.f, 0.f, 0.f, 0.f, 0.f, 0.f, 0.f, 0.f};

#pragma unroll
        for (int c = 0; c < 3; ++c) {        // K chunks of 32 (80 padded to 96)
            v16h a;
#pragma unroll
            for (int h = 0; h < 16; ++h) {
                int k = c * 32 + ((h < 8) ? (koff + h) : (16 + koff + (h - 8)));
                a[h] = (_Float16)sh_desc[i][M][k];   // zeros for k>=80 (padded)
            }
#pragma unroll
            for (int nt = 0; nt < 5; ++nt) {
                int l = nt * 16 + M;
                v16h bf;
#pragma unroll
                for (int h = 0; h < 16; ++h) {
                    int k = c * 32 + ((h < 8) ? (koff + h) : (16 + koff + (h - 8)));
                    float w = (k < KDIM) ? W[((size_t)i * KDIM + k) * KDIM + l] : 0.0f;
                    bf[h] = (_Float16)w;
                }
                acc[nt] = __builtin_amdgcn_wmma_f32_16x16x32_f16(
                    /*neg_a=*/false, a, /*neg_b=*/false, bf,
                    /*c_mod=*/(short)0, acc[nt],
                    /*reuse_a=*/false, /*reuse_b=*/false);
            }
        }

        // max over 16 rotations: 8 C-VGPRs (M=0..7 | 8..15) then lane-16 half swap
#pragma unroll
        for (int nt = 0; nt < 5; ++nt) {
            float m = acc[nt][0];
#pragma unroll
            for (int j = 1; j < 8; ++j) m = fmaxf(m, acc[nt][j]);
            m = fmaxf(m, __shfl_xor(m, 16, 32));
            int l = nt * 16 + M;
            float val = fmaxf(m + bconv[i * KDIM + l], 0.0f);
            if (lane < 16)
                out[(size_t)s * (KDIM * N_FEAT) + l * N_FEAT + i] = val;
        }
    }
}

extern "C" void kernel_launch(void* const* d_in, const int* in_sizes, int n_in,
                              void* d_out, int out_size, void* d_ws, size_t ws_size,
                              hipStream_t stream) {
    const float* x           = (const float*)d_in[0];
    const float* mu_rho      = (const float*)d_in[1];
    const float* sigma_rho   = (const float*)d_in[2];
    const float* mu_theta    = (const float*)d_in[3];
    const float* sigma_theta = (const float*)d_in[4];
    const float* W_conv      = (const float*)d_in[5];
    const float* b_conv      = (const float*)d_in[6];
    float* out = (float*)d_out;

    conv_fused_kernel<<<NS, 256, 0, stream>>>(x, mu_rho, sigma_rho, mu_theta,
                                              sigma_theta, W_conv, b_conv, out);
}
